// MergedW8A8Linear_50405736186187
// MI455X (gfx1250) — 8-bit matrix operand order repaired and checked by reading, compiled, not run on hardware
//
#include <hip/hip_runtime.h>
#include <stdint.h>


#define MM 64
#define KK 8192
#define NN 10240

typedef __attribute__((ext_vector_type(8)))  int       v8i;
typedef __attribute__((ext_vector_type(8)))  _Float16  h8;
typedef __attribute__((ext_vector_type(8)))  char      c8;

__global__ __launch_bounds__(256) void quantize_x_kernel(
    const _Float16* __restrict__ x,
    int8_t* __restrict__ xq,
    float* __restrict__ x_scale)
{
    const int m = blockIdx.x;
    const int t = threadIdx.x;

    const h8* __restrict__ rp = (const h8*)(x + (size_t)m * KK);
    float fv[32];
    float mx = 0.0f;

    #pragma unroll
    for (int c = 0; c < 4; ++c) {
        h8 hv = rp[c * 256 + t];
        #pragma unroll
        for (int j = 0; j < 8; ++j) {
            float f = (float)hv[j];
            fv[c * 8 + j] = f;
            mx = fmaxf(mx, fabsf(f));
        }
    }

    __shared__ float red[256];
    red[t] = mx;
    __syncthreads();
    #pragma unroll
    for (int s = 128; s > 0; s >>= 1) {
        if (t < s) red[t] = fmaxf(red[t], red[t + s]);
        __syncthreads();
    }
    const float maxv = red[0];
    const float inv  = (maxv > 0.0f) ? (127.0f / maxv) : 0.0f;

    if (t == 0) x_scale[m] = maxv * (1.0f / 127.0f);

    c8* __restrict__ qp = (c8*)(xq + (size_t)m * KK);
    #pragma unroll
    for (int c = 0; c < 4; ++c) {
        c8 q;
        #pragma unroll
        for (int j = 0; j < 8; ++j) {
            float v = fv[c * 8 + j] * inv;
            v = fminf(fmaxf(__builtin_rintf(v), -127.0f), 127.0f);
            q[j] = (char)(int)v;
        }
        qp[c * 256 + t] = q;
    }
}

__global__ __launch_bounds__(256) void w8a8_gemm_kernel(
    const int8_t*  __restrict__ xq,
    const float*   __restrict__ x_scale,
    const int8_t*  __restrict__ weight,
    const float*   __restrict__ w_scale,
    const _Float16* __restrict__ bias,
    _Float16*      __restrict__ out)
{
    const int lane = threadIdx.x & 31;
    const int wave = blockIdx.x * (blockDim.x >> 5) + (threadIdx.x >> 5);
    const int n_base = wave * 32;
    const int lo = lane & 15;
    const int hi = lane >> 4;

    const int kA = hi * 16;
    const int kB = hi * 16;

    const int8_t* __restrict__ wrow0 = weight + (size_t)(n_base + lo)      * KK;
    const int8_t* __restrict__ wrow1 = weight + (size_t)(n_base + 16 + lo) * KK;
    const int8_t* __restrict__ xrow0 = xq + (size_t)(lo)      * KK;
    const int8_t* __restrict__ xrow1 = xq + (size_t)(16 + lo) * KK;
    const int8_t* __restrict__ xrow2 = xq + (size_t)(32 + lo) * KK;
    const int8_t* __restrict__ xrow3 = xq + (size_t)(48 + lo) * KK;

    v8i acc[2][4];
    #pragma unroll
    for (int nt = 0; nt < 2; ++nt)
        #pragma unroll
        for (int mt = 0; mt < 4; ++mt)
            acc[nt][mt] = (v8i)0;

    union A8 { int2 d[4]; v8i v; };
    union B8 { int4 q[2]; v8i v; };

    for (int k = 0; k < KK; k += 64) {
        __builtin_prefetch(wrow0 + k + 1024, 0, 1);
        __builtin_prefetch(wrow1 + k + 1024, 0, 1);

        B8 b0, b1, b2, b3;
        b0.q[0] = *(const int4*)(xrow0 + k + kB);
        b0.q[1] = *(const int4*)(xrow0 + k + kB + 32);
        b1.q[0] = *(const int4*)(xrow1 + k + kB);
        b1.q[1] = *(const int4*)(xrow1 + k + kB + 32);
        b2.q[0] = *(const int4*)(xrow2 + k + kB);
        b2.q[1] = *(const int4*)(xrow2 + k + kB + 32);
        b3.q[0] = *(const int4*)(xrow3 + k + kB);
        b3.q[1] = *(const int4*)(xrow3 + k + kB + 32);

        A8 a0, a1;
        a0.d[0] = *(const int2*)(wrow0 + k + kA);
        a0.d[1] = *(const int2*)(wrow0 + k + kA + 8);
        a0.d[2] = *(const int2*)(wrow0 + k + kA + 32);
        a0.d[3] = *(const int2*)(wrow0 + k + kA + 40);
        a1.d[0] = *(const int2*)(wrow1 + k + kA);
        a1.d[1] = *(const int2*)(wrow1 + k + kA + 8);
        a1.d[2] = *(const int2*)(wrow1 + k + kA + 32);
        a1.d[3] = *(const int2*)(wrow1 + k + kA + 40);

        acc[0][0] = __builtin_amdgcn_wmma_i32_16x16x64_iu8(true, a0.v, true, b0.v, acc[0][0], false, false);
        acc[0][1] = __builtin_amdgcn_wmma_i32_16x16x64_iu8(true, a0.v, true, b1.v, acc[0][1], false, false);
        acc[0][2] = __builtin_amdgcn_wmma_i32_16x16x64_iu8(true, a0.v, true, b2.v, acc[0][2], false, false);
        acc[0][3] = __builtin_amdgcn_wmma_i32_16x16x64_iu8(true, a0.v, true, b3.v, acc[0][3], false, false);
        acc[1][0] = __builtin_amdgcn_wmma_i32_16x16x64_iu8(true, a1.v, true, b0.v, acc[1][0], false, false);
        acc[1][1] = __builtin_amdgcn_wmma_i32_16x16x64_iu8(true, a1.v, true, b1.v, acc[1][1], false, false);
        acc[1][2] = __builtin_amdgcn_wmma_i32_16x16x64_iu8(true, a1.v, true, b2.v, acc[1][2], false, false);
        acc[1][3] = __builtin_amdgcn_wmma_i32_16x16x64_iu8(true, a1.v, true, b3.v, acc[1][3], false, false);
    }

    #pragma unroll
    for (int nt = 0; nt < 2; ++nt) {
        const int n0 = n_base + nt * 16 + hi * 8;
        const float4 s0 = *(const float4*)(w_scale + n0);
        const float4 s1 = *(const float4*)(w_scale + n0 + 4);
        const float sc[8] = {s0.x, s0.y, s0.z, s0.w, s1.x, s1.y, s1.z, s1.w};
        const h8 bv = *(const h8*)(bias + n0);

        #pragma unroll
        for (int mt = 0; mt < 4; ++mt) {
            const int m = mt * 16 + lo;
            const float xs = x_scale[m];
            h8 r;
            #pragma unroll
            for (int v = 0; v < 8; ++v) {
                float f = (float)acc[nt][mt][v] * xs * sc[v] + (float)bv[v];
                r[v] = (_Float16)f;
            }
            *(h8*)(out + (size_t)m * NN + n0) = r;
        }
    }
}

extern "C" void kernel_launch(void* const* d_in, const int* in_sizes, int n_in,
                              void* d_out, int out_size, void* d_ws, size_t ws_size,
                              hipStream_t stream)
{
    const _Float16* x      = (const _Float16*)d_in[0];
    const int8_t*   weight = (const int8_t*)d_in[1];
    const float*    scale  = (const float*)d_in[2];
    const _Float16* bias   = (const _Float16*)d_in[3];
    _Float16*       out    = (_Float16*)d_out;

    int8_t* xq      = (int8_t*)d_ws;
    float*  x_scale = (float*)((char*)d_ws + (size_t)MM * KK);

    quantize_x_kernel<<<MM, 256, 0, stream>>>(x, xq, x_scale);

    w8a8_gemm_kernel<<<NN / 32 / 8, 256, 0, stream>>>(
        xq, x_scale, weight, scale, bias, out);
}
